// AdvancedTransformerBlock_25623774888465
// MI455X (gfx1250) — matrix operand order repaired and checked by reading, not compiled, not run on hardware
//
#include <hip/hip_runtime.h>
#include <hip/hip_bf16.h>
#include <stdint.h>

// ---------------- CDNA5 WMMA types ----------------
typedef __attribute__((ext_vector_type(16))) __bf16 v16bf;
typedef __attribute__((ext_vector_type(8)))  __bf16 v8bf;
typedef __attribute__((ext_vector_type(8)))  float  v8f;

__device__ __forceinline__ v16bf frag_cat(v8bf lo, v8bf hi) {
  return __builtin_shufflevector(lo, hi, 0,1,2,3,4,5,6,7,8,9,10,11,12,13,14,15);
}

// Async DMA: global -> LDS, 16 bytes per lane, tracked by ASYNCcnt.
__device__ __forceinline__ void async_copy16(unsigned lds_off, const void* gptr) {
  asm volatile("global_load_async_to_lds_b128 %0, %1, off"
               :: "v"(lds_off), "v"((unsigned long long)(uintptr_t)gptr)
               : "memory");
}
__device__ __forceinline__ void wait_async0() {
  asm volatile("s_wait_asynccnt 0x0" ::: "memory");
}

// =====================================================================
// bf16 WMMA GEMM:  out = ACT( A[M,K] * W[N,K]^T + bias + acc )
//  - block tile 128x128, 8 wave32 (wave tile 32x64 -> 8 WMMA / K-step)
//  - A and B tiles staged via GLOBAL_LOAD_ASYNC_TO_LDS_B128, double-buffered
//  - fragments follow ISA 7.12.2 VGPR layouts
// =====================================================================
template<int ACT>
__global__ __launch_bounds__(256)
void gemm_wmma_bf16(const __hip_bfloat16* __restrict__ A, int lda,
                    const __hip_bfloat16* __restrict__ W, int ldb,
                    const float* __restrict__ bias,
                    const float* __restrict__ accbuf,
                    float* __restrict__ outf,
                    __hip_bfloat16* __restrict__ outb,
                    int ldc, int K)
{
  __shared__ alignas(16) __bf16 As[2][128][40];  // 32 K-halves + pad
  __shared__ alignas(16) __bf16 Bs[2][128][40];

  const int tid  = threadIdx.x;
  const int lane = tid & 31;
  const int wave = tid >> 5;
  const int wm   = wave & 3;          // 4 M-quadrants of 32 rows
  const int wn   = wave >> 2;         // 2 N-halves  of 64 cols
  const int m0   = blockIdx.y * 128;
  const int n0   = blockIdx.x * 128;
  const int lrow = lane & 15;
  const int hl   = lane >> 4;         // half-wave select

  const __bf16* Ab = (const __bf16*)A;
  const __bf16* Wb = (const __bf16*)W;

  const int srow   = tid >> 1;        // 0..127 staged row/col
  const int schunk = (tid & 1) * 16;  // 0 or 16 halves

  auto stage = [&](int buf, int k0) {
    const __bf16* asrc = Ab + (size_t)(m0 + srow) * lda + k0 + schunk;
    const __bf16* bsrc = Wb + (size_t)(n0 + srow) * ldb + k0 + schunk;
    const unsigned la = (unsigned)(uintptr_t)&As[buf][srow][schunk];
    const unsigned lb = (unsigned)(uintptr_t)&Bs[buf][srow][schunk];
    async_copy16(la,      asrc);
    async_copy16(la + 16, asrc + 8);
    async_copy16(lb,      bsrc);
    async_copy16(lb + 16, bsrc + 8);
  };

  v8f c[2][4] = {};
  stage(0, 0);
  int buf = 0;

  for (int k0 = 0; k0 < K; k0 += 32) {
    wait_async0();       // my async writes for `buf` complete
    __syncthreads();     // everyone's writes visible / prev reads done
    if (k0 + 32 < K) stage(buf ^ 1, k0 + 32);   // DMA next tile during compute

    // ---- A fragments (16-bit A 16x32 layout: K chunks {0-7,16-23}/{8-15,24-31}) ----
    v16bf a[2], b[4];
    #pragma unroll
    for (int mi = 0; mi < 2; ++mi) {
      const int ar = wm * 32 + mi * 16 + lrow;
      a[mi] = frag_cat(*(const v8bf*)&As[buf][ar][hl * 8],
                       *(const v8bf*)&As[buf][ar][hl * 8 + 16]);
    }
    // ---- B fragments (B 32x16: lane = N col, K chunks {0-7,16-23}/{8-15,24-31}) ----
    #pragma unroll
    for (int ni = 0; ni < 4; ++ni) {
      const int bc = wn * 64 + ni * 16 + lrow;
      b[ni] = frag_cat(*(const v8bf*)&Bs[buf][bc][hl * 8],
                       *(const v8bf*)&Bs[buf][bc][hl * 8 + 16]);
    }
    // ---- 8x v_wmma_f32_16x16x32_bf16 ----
    #pragma unroll
    for (int mi = 0; mi < 2; ++mi)
      #pragma unroll
      for (int ni = 0; ni < 4; ++ni)
        c[mi][ni] = __builtin_amdgcn_wmma_f32_16x16x32_bf16(
            false, a[mi], false, b[ni], (short)0, c[mi][ni], false, false);
    buf ^= 1;
  }

  // ---- epilogue (C layout: VGPR e -> row e / e+8 per half-wave) ----
  #pragma unroll
  for (int mi = 0; mi < 2; ++mi) {
    #pragma unroll
    for (int ni = 0; ni < 4; ++ni) {
      const int col = n0 + wn * 64 + ni * 16 + lrow;
      const float bv = bias ? bias[col] : 0.0f;
      #pragma unroll
      for (int e = 0; e < 8; ++e) {
        const int row = m0 + wm * 32 + mi * 16 + hl * 8 + e;
        const size_t o = (size_t)row * ldc + col;
        float v = c[mi][ni][e] + bv;
        if (accbuf) v += accbuf[o];
        if      (ACT == 1) v = 1.0f / (1.0f + __expf(-v));                        // sigmoid
        else if (ACT == 2) v = tanhf(v);                                          // tanh
        else if (ACT == 3) v = 0.5f * v * (1.0f + erff(v * 0.70710678118654752f));// exact GELU
        if (outf) outf[o] = v;
        if (outb) outb[o] = __float2bfloat16(v);
      }
    }
  }
}

// =====================================================================
// LayerNorm (f32 in) -> bf16 out.  One row (1024) per 256-thread block.
// =====================================================================
__global__ __launch_bounds__(256)
void ln_bf16_kernel(const float* __restrict__ x,
                    const float* __restrict__ w,
                    const float* __restrict__ b,
                    __hip_bfloat16* __restrict__ out)
{
  __shared__ float s1[256], s2[256];
  const int row = blockIdx.x;
  const float* xr = x + (size_t)row * 1024;
  float a = 0.f, q = 0.f;
  for (int i = threadIdx.x; i < 1024; i += 256) { float v = xr[i]; a += v; q += v * v; }
  s1[threadIdx.x] = a; s2[threadIdx.x] = q;
  __syncthreads();
  for (int st = 128; st > 0; st >>= 1) {
    if (threadIdx.x < st) { s1[threadIdx.x] += s1[threadIdx.x + st];
                            s2[threadIdx.x] += s2[threadIdx.x + st]; }
    __syncthreads();
  }
  const float mu  = s1[0] * (1.0f / 1024.0f);
  const float var = s2[0] * (1.0f / 1024.0f) - mu * mu;
  const float inv = rsqrtf(var + 1e-5f);
  __hip_bfloat16* orow = out + (size_t)row * 1024;
  for (int i = threadIdx.x; i < 1024; i += 256)
    orow[i] = __float2bfloat16((xr[i] - mu) * inv * w[i] + b[i]);
}

// =====================================================================
// Attention per (batch, head): L=60, hd=64.  K/V/scores in LDS, f32 math.
// qkv rows: [ q(1024) | k(1024) | v(1024) ]  (bf16)
// =====================================================================
__global__ __launch_bounds__(64)
void attn_kernel(const __hip_bfloat16* __restrict__ qkv,
                 __hip_bfloat16* __restrict__ ctx, int causal)
{
  __shared__ float Ks[60][64];
  __shared__ float Vs[60][64];
  __shared__ float Ps[64][61];   // padded: conflict-free
  const int bh = blockIdx.x;
  const int bi = bh >> 4, h = bh & 15;
  const size_t base = (size_t)bi * 60 * 3072;

  for (int i = threadIdx.x; i < 60 * 64; i += 64) {
    const int l = i >> 6, d = i & 63;
    Ks[l][d] = __bfloat162float(qkv[base + (size_t)l * 3072 + 1024 + h * 64 + d]);
    Vs[l][d] = __bfloat162float(qkv[base + (size_t)l * 3072 + 2048 + h * 64 + d]);
  }
  __syncthreads();

  const int l = threadIdx.x;
  if (l < 60) {
    float qv[64];
    #pragma unroll
    for (int d = 0; d < 64; ++d)
      qv[d] = __bfloat162float(qkv[base + (size_t)l * 3072 + h * 64 + d]) * 0.125f;

    const int jmax = causal ? (l + 1) : 60;
    float m = -3.402823466e38f;
    for (int j = 0; j < jmax; ++j) {
      float s = 0.f;
      #pragma unroll
      for (int d = 0; d < 64; ++d) s += qv[d] * Ks[j][d];
      Ps[l][j] = s; m = fmaxf(m, s);
    }
    float ssum = 0.f;
    for (int j = 0; j < jmax; ++j) { float p = __expf(Ps[l][j] - m); Ps[l][j] = p; ssum += p; }
    const float inv = 1.0f / ssum;

    for (int d0 = 0; d0 < 64; d0 += 8) {
      float acc[8] = {0,0,0,0,0,0,0,0};
      for (int j = 0; j < jmax; ++j) {
        const float p = Ps[l][j];
        #pragma unroll
        for (int dd = 0; dd < 8; ++dd) acc[dd] += p * Vs[j][d0 + dd];
      }
      const size_t o = ((size_t)bi * 60 + l) * 1024 + h * 64 + d0;
      #pragma unroll
      for (int dd = 0; dd < 8; ++dd) ctx[o + dd] = __float2bfloat16(acc[dd] * inv);
    }
  }
}

// ---------------- elementwise kernels ----------------
__global__ void conv_f32_bf16(const float* __restrict__ s, __hip_bfloat16* __restrict__ d, int n) {
  for (int i = blockIdx.x * blockDim.x + threadIdx.x; i < n; i += gridDim.x * blockDim.x)
    d[i] = __float2bfloat16(s[i]);
}
__global__ void gate_combine(const float* __restrict__ x, const float* __restrict__ g,
                             const float* __restrict__ so, const float* __restrict__ lo,
                             float* __restrict__ x1f, __hip_bfloat16* __restrict__ x1b, int n) {
  for (int i = blockIdx.x * blockDim.x + threadIdx.x; i < n; i += gridDim.x * blockDim.x) {
    const float gv = g[i];
    const float v = x[i] + gv * so[i] + (1.0f - gv) * lo[i];
    x1f[i] = v; x1b[i] = __float2bfloat16(v);
  }
}
__global__ void mul_rx(const float* __restrict__ r, const float* __restrict__ x1f,
                       __hip_bfloat16* __restrict__ rx, int n) {
  for (int i = blockIdx.x * blockDim.x + threadIdx.x; i < n; i += gridDim.x * blockDim.x)
    rx[i] = __float2bfloat16(r[i] * x1f[i]);
}
__global__ void final_fuse(const float* __restrict__ x1f, const float* __restrict__ z,
                           const float* __restrict__ ht, float* __restrict__ out, int n) {
  for (int i = blockIdx.x * blockDim.x + threadIdx.x; i < n; i += gridDim.x * blockDim.x) {
    const float zv = z[i];
    out[i] = (1.0f - zv) * x1f[i] + zv * ht[i];
  }
}

// =====================================================================
// Host orchestration
// =====================================================================
extern "C" void kernel_launch(void* const* d_in, const int* in_sizes, int n_in,
                              void* d_out, int out_size, void* d_ws, size_t ws_size,
                              hipStream_t stream)
{
  (void)in_sizes; (void)n_in; (void)out_size; (void)ws_size;
  const int D = 1024;
  const size_t M  = 30720;          // B*L = 512*60
  const size_t MD = M * D;

  const float* x        = (const float*)d_in[0];
  const float* ln1_w    = (const float*)d_in[1];
  const float* ln1_b    = (const float*)d_in[2];
  const float* s_wqkv   = (const float*)d_in[3];
  const float* s_bqkv   = (const float*)d_in[4];
  const float* s_wo     = (const float*)d_in[5];
  const float* s_bo     = (const float*)d_in[6];
  const float* l_wqkv   = (const float*)d_in[7];
  const float* l_bqkv   = (const float*)d_in[8];
  const float* l_wo     = (const float*)d_in[9];
  const float* l_bo     = (const float*)d_in[10];
  const float* gate_w   = (const float*)d_in[11];
  const float* gate_b   = (const float*)d_in[12];
  const float* ln2_w    = (const float*)d_in[13];
  const float* ln2_b    = (const float*)d_in[14];
  const float* ffn_w1   = (const float*)d_in[15];
  const float* ffn_b1   = (const float*)d_in[16];
  const float* ffn_w2   = (const float*)d_in[17];
  const float* ffn_b2   = (const float*)d_in[18];
  const float* r_w      = (const float*)d_in[19];
  const float* r_b      = (const float*)d_in[20];
  const float* z_w      = (const float*)d_in[21];
  const float* z_b      = (const float*)d_in[22];
  const float* c_w      = (const float*)d_in[23];
  const float* c_b      = (const float*)d_in[24];
  float* out = (float*)d_out;

  uint8_t* ws = (uint8_t*)d_ws;
  size_t off = 0;
  auto alloc = [&](size_t bytes) -> void* {
    void* p = ws + off; off += (bytes + 255) & ~(size_t)255; return p;
  };
  typedef __hip_bfloat16 bf;

  // bf16 weight copies
  bf* wb_qkv_s = (bf*)alloc((size_t)3*D*D*2);
  bf* wb_qkv_l = (bf*)alloc((size_t)3*D*D*2);
  bf* wb_wo_s  = (bf*)alloc((size_t)D*D*2);
  bf* wb_wo_l  = (bf*)alloc((size_t)D*D*2);
  bf* wb_gate  = (bf*)alloc((size_t)2*D*D*2);
  bf* wb_f1    = (bf*)alloc((size_t)2*D*D*2);
  bf* wb_f2    = (bf*)alloc((size_t)2*D*D*2);
  bf* wb_r     = (bf*)alloc((size_t)2*D*D*2);
  bf* wb_z     = (bf*)alloc((size_t)2*D*D*2);
  bf* wb_c     = (bf*)alloc((size_t)2*D*D*2);

  // activation regions (lifetime-based reuse)
  bf*  R_xn  = (bf*)alloc(MD * 2);             // xn_bf; later h_bf
  uint8_t* R_qs = (uint8_t*)alloc(MD * 3 * 2); // qkv_s; later so_f32|so_bf; later htilde
  uint8_t* R_ql = (uint8_t*)alloc(MD * 3 * 2); // qkv_l; later lo_f32|lo_bf
  bf*  R_cs  = (bf*)alloc(MD * 2);             // ctx_s; later rx_bf
  bf*  R_cl  = (bf*)alloc(MD * 2);             // ctx_l; later ffn_bf
  float* R_g = (float*)alloc(MD * 4);          // g; later r
  float* R_x1f = (float*)alloc(MD * 4);        // x1 f32
  bf*  R_x1b = (bf*)alloc(MD * 2);             // x1 bf16
  uint8_t* R_hf = (uint8_t*)alloc(MD * 2 * 2); // gelu(h) bf16 [M,2048]; later z f32

  bf*    qkv_s = (bf*)R_qs;
  bf*    qkv_l = (bf*)R_ql;
  float* so_f  = (float*)R_qs;  bf* so_b = (bf*)(R_qs + MD * 4);
  float* lo_f  = (float*)R_ql;  bf* lo_b = (bf*)(R_ql + MD * 4);
  float* ht    = (float*)R_qs;
  float* rbuf  = R_g;
  bf*    rx    = R_cs;
  bf*    ffn_o = R_cl;
  bf*    h_bf  = R_xn;
  bf*    hf_bf = (bf*)R_hf;
  float* zbuf  = (float*)R_hf;

  const dim3 blk(256);
  const int EW_GRID = 8192;
  const int nMD = (int)MD;

  auto gemm = [&](const bf* Ain, int lda, const bf* Win, int ldb, const float* bias,
                  const float* acc, float* outf, bf* outb, int ldc, int N, int K, int act) {
    dim3 g(N / 128, (unsigned)(M / 128));
    switch (act) {
      case 1: gemm_wmma_bf16<1><<<g, blk, 0, stream>>>(Ain, lda, Win, ldb, bias, acc, outf, outb, ldc, K); break;
      case 2: gemm_wmma_bf16<2><<<g, blk, 0, stream>>>(Ain, lda, Win, ldb, bias, acc, outf, outb, ldc, K); break;
      case 3: gemm_wmma_bf16<3><<<g, blk, 0, stream>>>(Ain, lda, Win, ldb, bias, acc, outf, outb, ldc, K); break;
      default: gemm_wmma_bf16<0><<<g, blk, 0, stream>>>(Ain, lda, Win, ldb, bias, acc, outf, outb, ldc, K); break;
    }
  };

  // ---- weight conversion fp32 -> bf16 ----
  conv_f32_bf16<<<4096, blk, 0, stream>>>(s_wqkv, wb_qkv_s, 3*D*D);
  conv_f32_bf16<<<4096, blk, 0, stream>>>(l_wqkv, wb_qkv_l, 3*D*D);
  conv_f32_bf16<<<2048, blk, 0, stream>>>(s_wo,   wb_wo_s,  D*D);
  conv_f32_bf16<<<2048, blk, 0, stream>>>(l_wo,   wb_wo_l,  D*D);
  conv_f32_bf16<<<4096, blk, 0, stream>>>(gate_w, wb_gate,  2*D*D);
  conv_f32_bf16<<<4096, blk, 0, stream>>>(ffn_w1, wb_f1,    2*D*D);
  conv_f32_bf16<<<4096, blk, 0, stream>>>(ffn_w2, wb_f2,    2*D*D);
  conv_f32_bf16<<<4096, blk, 0, stream>>>(r_w,    wb_r,     2*D*D);
  conv_f32_bf16<<<4096, blk, 0, stream>>>(z_w,    wb_z,     2*D*D);
  conv_f32_bf16<<<4096, blk, 0, stream>>>(c_w,    wb_c,     2*D*D);

  // ---- 1. ln1 ----
  ln_bf16_kernel<<<(int)M, blk, 0, stream>>>(x, ln1_w, ln1_b, R_xn);

  // ---- 2. QKV GEMMs (bf16 out) ----
  gemm(R_xn, D, wb_qkv_s, D, s_bqkv, nullptr, nullptr, qkv_s, 3*D, 3*D, D, 0);
  gemm(R_xn, D, wb_qkv_l, D, l_bqkv, nullptr, nullptr, qkv_l, 3*D, 3*D, D, 0);

  // ---- 3. attention ----
  attn_kernel<<<512 * 16, dim3(64), 0, stream>>>(qkv_s, R_cs, 1);
  attn_kernel<<<512 * 16, dim3(64), 0, stream>>>(qkv_l, R_cl, 0);

  // ---- 4. out projections (f32 + bf16 out) ----
  gemm(R_cs, D, wb_wo_s, D, s_bo, nullptr, so_f, so_b, D, D, D, 0);
  gemm(R_cl, D, wb_wo_l, D, l_bo, nullptr, lo_f, lo_b, D, D, D, 0);

  // ---- 5. gate = sigmoid([short||long] @ gate_w^T + b)  (split concat) ----
  gemm(so_b, D, wb_gate,     2*D, nullptr, nullptr, R_g, nullptr, D, D, D, 0);
  gemm(lo_b, D, wb_gate + D, 2*D, gate_b,  R_g,     R_g, nullptr, D, D, D, 1);

  // ---- 6. x1 = x + g*short + (1-g)*long ----
  gate_combine<<<EW_GRID, blk, 0, stream>>>(x, R_g, so_f, lo_f, R_x1f, R_x1b, nMD);

  // ---- 7. ln2 + FFN ----
  ln_bf16_kernel<<<(int)M, blk, 0, stream>>>(R_x1f, ln2_w, ln2_b, h_bf);
  gemm(h_bf, D, wb_f1, D, ffn_b1, nullptr, nullptr, hf_bf, 2*D, 2*D, D, 3);   // GELU
  gemm(hf_bf, 2*D, wb_f2, 2*D, ffn_b2, nullptr, nullptr, ffn_o, D, D, 2*D, 0);

  // ---- 8. r = sigmoid([x1||ffn] @ r_w^T + b) ----
  gemm(R_x1b, D, wb_r,     2*D, nullptr, nullptr, rbuf, nullptr, D, D, D, 0);
  gemm(ffn_o, D, wb_r + D, 2*D, r_b,     rbuf,    rbuf, nullptr, D, D, D, 1);
  mul_rx<<<EW_GRID, blk, 0, stream>>>(rbuf, R_x1f, rx, nMD);

  // ---- 9. z = sigmoid([x1||ffn] @ z_w^T + b) ----
  gemm(R_x1b, D, wb_z,     2*D, nullptr, nullptr, zbuf, nullptr, D, D, D, 0);
  gemm(ffn_o, D, wb_z + D, 2*D, z_b,     zbuf,    zbuf, nullptr, D, D, D, 1);

  // ---- 10. h_tilde = tanh([r*x1||ffn] @ c_w^T + b) ----
  gemm(rx,    D, wb_c,     2*D, nullptr, nullptr, ht, nullptr, D, D, D, 0);
  gemm(ffn_o, D, wb_c + D, 2*D, c_b,     ht,      ht, nullptr, D, D, D, 2);

  // ---- 11. out = (1-z)*x1 + z*h_tilde ----
  final_fuse<<<EW_GRID, blk, 0, stream>>>(R_x1f, zbuf, ht, out, nMD);
}
